// EfficientViTJointBlock_36764920054577
// MI455X (gfx1250) — matrix operand order repaired and checked by reading, not compiled, not run on hardware
//
#include <hip/hip_runtime.h>
#include <hip/hip_bf16.h>
#include <math.h>
#include <stdint.h>

typedef __attribute__((ext_vector_type(16))) _Float16 v16h;
typedef __attribute__((ext_vector_type(8)))  _Float16 v8h;
typedef __attribute__((ext_vector_type(8)))  float    v8f;

#define CAT16(lo, hi) __builtin_shufflevector(lo, hi, 0,1,2,3,4,5,6,7,8,9,10,11,12,13,14,15)

#define BSZ   16
#define CCH   256
#define HH    64
#define WW    64
#define HWSP  4096            // H*W
#define NHD   4               // heads
#define HDD   64              // head dim
#define HIDC  1024
#define EPSF  1e-6f
#define CHW   ((long)CCH * HWSP)          // per-batch elems of a C-channel tensor

// ---------------------------------------------------------------------------
// LayerNorm2d over channels at each (b, h, w). Optional fused residual input.
// ---------------------------------------------------------------------------
__global__ __launch_bounds__(256) void ln2d_k(const float* __restrict__ in,
                                              const float* __restrict__ res,
                                              const float* __restrict__ g,
                                              const float* __restrict__ be,
                                              float* __restrict__ out) {
  long p  = (long)blockIdx.x * blockDim.x + threadIdx.x;   // 0 .. B*HW-1
  long bb = p / HWSP, n = p % HWSP;
  const float* ip = in + bb * CHW + n;
  const float* rp = res ? res + bb * CHW + n : nullptr;
  float s = 0.f, sq = 0.f;
  for (int c = 0; c < CCH; ++c) {
    float v = ip[(long)c * HWSP];
    if (rp) v += rp[(long)c * HWSP];
    s += v; sq += v * v;
  }
  const float m   = s * (1.f / CCH);
  const float var = sq * (1.f / CCH) - m * m;
  const float inv = rsqrtf(var + EPSF);
  float* op = out + bb * CHW + n;
  for (int c = 0; c < CCH; ++c) {
    float v = ip[(long)c * HWSP];
    if (rp) v += rp[(long)c * HWSP];
    op[(long)c * HWSP] = (v - m) * inv * g[c] + be[c];
  }
}

// ---------------------------------------------------------------------------
// 1x1 conv as WMMA GEMM:  C[b][m][n] = sum_k A[m][k] * B[b][k][n] + bias[m]
// (+ optional residual).  Block tile 64x128, BK = 32, 8 waves, 4 WMMAs per
// A-fragment.  Double-buffered LDS: next chunk's global loads are issued into
// registers before the WMMA phase, stored to the alternate buffer, ONE
// workgroup barrier per chunk.  B stream prefetched two chunks ahead.
// ---------------------------------------------------------------------------
__global__ __launch_bounds__(256) void gemm1x1_k(const float* __restrict__ Aw,
                                                 const float* __restrict__ Bm,
                                                 const float* __restrict__ bias,
                                                 const float* __restrict__ res,
                                                 float* __restrict__ Cm,
                                                 int M, int K, long bBS, long cBS) {
  __shared__ __align__(16) _Float16 As[2][64][40];    // [buf][m][k]
  __shared__ __align__(16) _Float16 Bt[2][128][40];   // [buf][n][k] (transposed)
  const int bb = blockIdx.z;
  const float* Bb = Bm + (long)bb * bBS;
  float*       Cb = Cm + (long)bb * cBS;
  const int m0 = blockIdx.y * 64, n0 = blockIdx.x * 128;
  const int t    = threadIdx.x;
  const int lane = t & 31, wave = t >> 5;
  const int wm = wave & 3;             // wave row tile (16 rows)
  const int wq = wave >> 2;            // 0/1 -> col tiles wq*4 .. wq*4+3
  const int lm = lane & 15, hi = lane >> 4;
  const int kb = hi * 16;
  const int rA = t >> 2, cA = (t & 3) * 8;    // A loader: 64 rows x 32 cols
  const int rB = t >> 3, cB = (t & 7) * 16;   // B loader: 32 rows x 128 cols
  v8f acc[4] = {{}, {}, {}, {}};
  float aReg[8], bReg[16];
  // prologue: chunk 0 -> regs -> buf 0
  {
    const float* sa = Aw + (long)(m0 + rA) * K + cA;
#pragma unroll
    for (int j = 0; j < 8; ++j) aReg[j] = sa[j];
    const float* sb = Bb + (long)rB * HWSP + (n0 + cB);
#pragma unroll
    for (int j = 0; j < 16; ++j) bReg[j] = sb[j];
  }
#pragma unroll
  for (int j = 0; j < 8; ++j) As[0][rA][cA + j] = (_Float16)aReg[j];
#pragma unroll
  for (int j = 0; j < 16; ++j) Bt[0][cB + j][rB] = (_Float16)bReg[j];
  __syncthreads();
  int cur = 0;
  for (int k0 = 0; k0 < K; k0 += 32) {
    const bool last = (k0 + 32 >= K);
    if (!last) {   // issue next chunk's global loads early (latency under WMMA)
      const float* sa = Aw + (long)(m0 + rA) * K + (k0 + 32 + cA);
#pragma unroll
      for (int j = 0; j < 8; ++j) aReg[j] = sa[j];
      const float* sb = Bb + (long)(k0 + 32 + rB) * HWSP + (n0 + cB);
#pragma unroll
      for (int j = 0; j < 16; ++j) bReg[j] = sb[j];
      if (k0 + 64 < K) __builtin_prefetch(sb + 32 * HWSP, 0, 1);  // 2 ahead -> GL2
    }
    // compute on buf[cur]
    v8h alo = *(const v8h*)&As[cur][wm * 16 + lm][kb];
    v8h ahi = *(const v8h*)&As[cur][wm * 16 + lm][kb + 8];
    v16h af = CAT16(alo, ahi);
#pragma unroll
    for (int u = 0; u < 4; ++u) {
      const _Float16* bp = &Bt[cur][(wq * 4 + u) * 16 + lm][hi * 16];
      v16h bf = CAT16(*(const v8h*)bp, *(const v8h*)(bp + 8));
      acc[u] = __builtin_amdgcn_wmma_f32_16x16x32_f16(false, af, false, bf, (short)0, acc[u], false, false);
    }
    if (last) break;
    const int nxt = cur ^ 1;   // writers target the buffer nobody is reading
#pragma unroll
    for (int j = 0; j < 8; ++j) As[nxt][rA][cA + j] = (_Float16)aReg[j];
#pragma unroll
    for (int j = 0; j < 16; ++j) Bt[nxt][cB + j][rB] = (_Float16)bReg[j];
    __syncthreads();           // single barrier per chunk
    cur = nxt;
  }
#pragma unroll
  for (int r = 0; r < 8; ++r) {
    const int   m  = m0 + wm * 16 + hi * 8 + r;
    const float bv = bias[m];
#pragma unroll
    for (int u = 0; u < 4; ++u) {
      const int  na = n0 + (wq * 4 + u) * 16 + lm;
      const long i0 = (long)m * HWSP + na;
      float o = acc[u][r] + bv;
      if (res) o += res[(long)bb * cBS + i0];
      Cb[i0] = o;
    }
  }
}

// ---------------------------------------------------------------------------
// Depthwise conv: out = dw3(in)+b3 [+ dw5(in)+b5] [then exact GELU].
// ---------------------------------------------------------------------------
__global__ __launch_bounds__(256) void dw_k(const float* __restrict__ in, long inBS,
                                            const float* __restrict__ w3, const float* __restrict__ b3,
                                            const float* __restrict__ w5, const float* __restrict__ b5,
                                            float* __restrict__ out, long outBS,
                                            int Cn, int has5, int gelu) {
  __shared__ float tile[20][20];
  const int z = blockIdx.z;
  const int c = z % Cn, bb = z / Cn;
  const float* ip = in + (long)bb * inBS + (long)c * HWSP;
  const int h0 = blockIdx.y * 16, w0 = blockIdx.x * 16;
  const int t = threadIdx.y * 16 + threadIdx.x;
  for (int l = t; l < 400; l += 256) {
    const int ty = l / 20, tx = l % 20;
    const int gh = h0 + ty - 2, gw = w0 + tx - 2;
    tile[ty][tx] = (gh >= 0 && gh < HH && gw >= 0 && gw < WW) ? ip[gh * WW + gw] : 0.f;
  }
  __syncthreads();
  const int cy = threadIdx.y + 2, cx = threadIdx.x + 2;
  float acc = b3[c];
  const float* W3 = w3 + c * 9;
#pragma unroll
  for (int i = 0; i < 3; ++i)
#pragma unroll
    for (int j = 0; j < 3; ++j)
      acc += tile[cy - 1 + i][cx - 1 + j] * W3[i * 3 + j];
  if (has5) {
    acc += b5[c];
    const float* W5 = w5 + c * 25;
#pragma unroll
    for (int i = 0; i < 5; ++i)
#pragma unroll
      for (int j = 0; j < 5; ++j)
        acc += tile[cy - 2 + i][cx - 2 + j] * W5[i * 5 + j];
  }
  if (gelu) acc = 0.5f * acc * (1.f + erff(acc * 0.70710678118654752f));
  out[(long)bb * outBS + (long)c * HWSP + (h0 + threadIdx.y) * WW + (w0 + threadIdx.x)] = acc;
}

// ---------------------------------------------------------------------------
// Linear attention phase 1 (per (b, head)):
//   kv[d][e] = sum_n relu(k[d][n]) * v[e][n]   (WMMA, K = 4096)
//   ksum[d]  = sum_n relu(k[d][n])             (fused into register loads)
// Double-buffered LDS, one barrier per 32-wide chunk.
// ---------------------------------------------------------------------------
__global__ __launch_bounds__(256) void attn_kv_k(const float* __restrict__ qkvd,
                                                 float* __restrict__ kvout,
                                                 float* __restrict__ ksumout) {
  __shared__ __align__(16) _Float16 Ks[2][64][40];   // [buf][d][nloc]
  __shared__ __align__(16) _Float16 Vt[2][64][40];   // [buf][e][nloc]
  __shared__ float red[256];
  const int bh = blockIdx.x;
  const int bb = bh >> 2, hh = bh & 3;
  const float* kp = qkvd + (long)bb * (3 * CHW) + 1 * CHW + (long)hh * (HDD * HWSP);
  const float* vp = qkvd + (long)bb * (3 * CHW) + 2 * CHW + (long)hh * (HDD * HWSP);
  const int t = threadIdx.x;
  const int lane = t & 31, wave = t >> 5;
  const int wm = wave & 3, wc = (wave >> 2) * 2;
  const int lm = lane & 15, hi = lane >> 4;
  const int kb = hi * 16;
  const int rA = t >> 2, cA = (t & 3) * 8;    // 64 rows x 32 cols loader
  float kpart = 0.f;
  float kReg[8], vReg[8];
  v8f acc0 = {}, acc1 = {};
  // prologue: chunk 0 (relu K, accumulate ksum once per chunk at load time)
  {
    const float* sk = kp + (long)rA * HWSP + cA;
    const float* sv = vp + (long)rA * HWSP + cA;
#pragma unroll
    for (int j = 0; j < 8; ++j) { kReg[j] = fmaxf(sk[j], 0.f); kpart += kReg[j]; vReg[j] = sv[j]; }
  }
#pragma unroll
  for (int j = 0; j < 8; ++j) { Ks[0][rA][cA + j] = (_Float16)kReg[j]; Vt[0][rA][cA + j] = (_Float16)vReg[j]; }
  __syncthreads();
  int cur = 0;
  for (int n0 = 0; n0 < HWSP; n0 += 32) {
    const bool last = (n0 + 32 >= HWSP);
    if (!last) {
      const float* sk = kp + (long)rA * HWSP + (n0 + 32 + cA);
      const float* sv = vp + (long)rA * HWSP + (n0 + 32 + cA);
#pragma unroll
      for (int j = 0; j < 8; ++j) { kReg[j] = fmaxf(sk[j], 0.f); kpart += kReg[j]; vReg[j] = sv[j]; }
      if (n0 + 64 < HWSP) { __builtin_prefetch(sk + 32, 0, 1); __builtin_prefetch(sv + 32, 0, 1); }
    }
    v8h alo = *(const v8h*)&Ks[cur][wm * 16 + lm][kb];
    v8h ahi = *(const v8h*)&Ks[cur][wm * 16 + lm][kb + 8];
    v16h af = CAT16(alo, ahi);
    const _Float16* bp0 = &Vt[cur][wc * 16 + lm][hi * 16];
    v16h bf0 = CAT16(*(const v8h*)bp0, *(const v8h*)(bp0 + 8));
    const _Float16* bp1 = &Vt[cur][wc * 16 + 16 + lm][hi * 16];
    v16h bf1 = CAT16(*(const v8h*)bp1, *(const v8h*)(bp1 + 8));
    acc0 = __builtin_amdgcn_wmma_f32_16x16x32_f16(false, af, false, bf0, (short)0, acc0, false, false);
    acc1 = __builtin_amdgcn_wmma_f32_16x16x32_f16(false, af, false, bf1, (short)0, acc1, false, false);
    if (last) break;
    const int nxt = cur ^ 1;
#pragma unroll
    for (int j = 0; j < 8; ++j) { Ks[nxt][rA][cA + j] = (_Float16)kReg[j]; Vt[nxt][rA][cA + j] = (_Float16)vReg[j]; }
    __syncthreads();
    cur = nxt;
  }
  red[t] = kpart;
  __syncthreads();
  if ((t & 3) == 0)
    ksumout[bh * 64 + rA] = red[t] + red[t + 1] + red[t + 2] + red[t + 3];
  float* kvb = kvout + bh * 4096;
#pragma unroll
  for (int r = 0; r < 8; ++r) {
    const int d = wm * 16 + hi * 8 + r;
    kvb[d * 64 + wc * 16 + lm]      = acc0[r];
    kvb[d * 64 + wc * 16 + 16 + lm] = acc1[r];
  }
}

// ---------------------------------------------------------------------------
// Linear attention phase 2: kv tile (16 KB) arrives via gfx1250 async
// global->LDS copies (ASYNCcnt); q column held in registers.
// ---------------------------------------------------------------------------
__global__ __launch_bounds__(256) void attn_apply_k(const float* __restrict__ qkvd,
                                                    const float* __restrict__ kv,
                                                    const float* __restrict__ ksum,
                                                    float* __restrict__ attn) {
  __shared__ __align__(16) float kvS[64][64];
  __shared__ float ksS[64];
  const int bh = blockIdx.y;
  const int bb = bh >> 2, hh = bh & 3;
  const int t = threadIdx.x;
  {
    const unsigned lds_base = (unsigned)(uintptr_t)(&kvS[0][0]) + (unsigned)t * 64u;
    const unsigned long long gbase =
        (unsigned long long)(uintptr_t)(kv + (long)bh * 4096) + (unsigned long long)t * 64ull;
#pragma unroll
    for (int i = 0; i < 4; ++i) {
      const unsigned           loff  = lds_base + (unsigned)(i * 16);
      const unsigned long long gaddr = gbase + (unsigned long long)(i * 16);
      asm volatile("global_load_async_to_lds_b128 %0, %1, off"
                   :: "v"(loff), "v"(gaddr) : "memory");
    }
  }
  if (t < 64) ksS[t] = ksum[bh * 64 + t];
  asm volatile("s_wait_asynccnt 0x0" ::: "memory");
  __syncthreads();
  const int n = blockIdx.x * 256 + t;
  const float* qp = qkvd + (long)bb * (3 * CHW) + (long)hh * (HDD * HWSP) + n;
  float qv[64];
  float den = 0.f;
#pragma unroll
  for (int d = 0; d < 64; ++d) {
    const float q = fmaxf(qp[(long)d * HWSP], 0.f);
    qv[d] = q;
    den += q * ksS[d];
  }
  const float inv = 1.f / fmaxf(den, EPSF);
  float* op = attn + (long)bb * CHW + (long)hh * (HDD * HWSP) + n;
#pragma unroll 4
  for (int e = 0; e < 64; ++e) {
    float a = 0.f;
#pragma unroll
    for (int d = 0; d < 64; ++d) a += qv[d] * kvS[d][e];
    op[(long)e * HWSP] = a * inv;
  }
}

__global__ __launch_bounds__(256) void add_k(const float* __restrict__ a,
                                             const float* __restrict__ b,
                                             float* __restrict__ o, long n) {
  long i  = (long)blockIdx.x * blockDim.x + threadIdx.x;
  long st = (long)gridDim.x * blockDim.x;
  for (; i < n; i += st) o[i] = a[i] + b[i];
}

// ---------------------------------------------------------------------------
extern "C" void kernel_launch(void* const* d_in, const int* in_sizes, int n_in,
                              void* d_out, int out_size, void* d_ws, size_t ws_size,
                              hipStream_t stream) {
  const float* x = (const float*)d_in[0];
  auto P = [&](int i) { return (const float*)d_in[i]; };
  const float *pre_g = P(1),  *pre_b = P(2),  *lon_g = P(3),  *lon_b = P(4);
  const float *gbn_g = P(5),  *gbn_b = P(6),  *frn_g = P(7),  *frn_b = P(8);
  const float *ffn_g = P(9),  *ffn_b = P(10), *on_g  = P(11), *on_b  = P(12);
  const float *lpw_w = P(13), *lpw_b = P(14), *ld3_w = P(15), *ld3_b = P(16);
  const float *ld5_w = P(17), *ld5_b = P(18), *lf_w  = P(19), *lf_b  = P(20);
  const float *qkv_w = P(21), *qkv_b = P(22);
  const float *dw3w[3] = {P(23), P(27), P(31)};
  const float *dw3b[3] = {P(24), P(28), P(32)};
  const float *dw5w[3] = {P(25), P(29), P(33)};
  const float *dw5b[3] = {P(26), P(30), P(34)};
  const float *go_w = P(35), *go_b = P(36), *fp_w = P(37), *fp_b = P(38);
  const float *fe_w = P(39), *fe_b = P(40), *fd3_w = P(41), *fd3_b = P(42);
  const float *fs_w = P(43), *fs_b = P(44);

  float* out = (float*)d_out;
  float* ws  = (float*)d_ws;
  const long T = (long)BSZ * CHW;            // 16,777,216 floats (64 MiB)
  float* W0  = ws;                            // T
  float* W1  = ws + T;                        // T
  float* W2  = ws + 2 * T;                    // T
  float* R   = ws + 3 * T;                    // 6T shared region (qkv / ffn chunks)
  float* kvb = ws + 9 * T;                    // 64 * 4096
  float* ksb = kvb + 64 * 4096;               // 64 * 64

  const dim3 blk(256);
  const dim3 dwblk(16, 16);

  // 1. xn = LN(x, pre_norm)                                  -> W0
  ln2d_k<<<256, blk, 0, stream>>>(x, nullptr, pre_g, pre_b, W0);
  // 2. seed = local_pw(xn)                                   -> W1
  gemm1x1_k<<<dim3(32, 4, 16), blk, 0, stream>>>(lpw_w, W0, lpw_b, nullptr, W1, 256, 256, CHW, CHW);
  // 3. localdw = dw3(seed)+dw5(seed)                         -> W2
  dw_k<<<dim3(4, 4, BSZ * CCH), dwblk, 0, stream>>>(W1, CHW, ld3_w, ld3_b, ld5_w, ld5_b, W2, CHW, CCH, 1, 0);
  // 4. local = local_fuse(localdw)                           -> W1
  gemm1x1_k<<<dim3(32, 4, 16), blk, 0, stream>>>(lf_w, W2, lf_b, nullptr, W1, 256, 256, CHW, CHW);
  // 5. local = LN(local) (in-place safe: thread-private columns)
  ln2d_k<<<256, blk, 0, stream>>>(W1, nullptr, lon_g, lon_b, W1);
  // 6. qkv = qkv_conv(xn)                                    -> R[0..3T)
  gemm1x1_k<<<dim3(32, 12, 16), blk, 0, stream>>>(qkv_w, W0, qkv_b, nullptr, R, 768, 256, CHW, 3 * CHW);
  // 7. q/k/v depthwise                                       -> R[3T..6T)
  for (int i = 0; i < 3; ++i)
    dw_k<<<dim3(4, 4, BSZ * CCH), dwblk, 0, stream>>>(R + (long)i * CHW, 3 * CHW,
                                                      dw3w[i], dw3b[i], dw5w[i], dw5b[i],
                                                      R + 3 * T + (long)i * CHW, 3 * CHW, CCH, 1, 0);
  // 8. kv / ksum per (b, head)  (WMMA over N=4096)
  attn_kv_k<<<64, blk, 0, stream>>>(R + 3 * T, kvb, ksb);
  // 9. attn = apply(q, kv, ksum)                             -> R[0..T)
  attn_apply_k<<<dim3(16, 64), blk, 0, stream>>>(R + 3 * T, kvb, ksb, R);
  // 10. glob = global_out(attn)                              -> W0
  gemm1x1_k<<<dim3(32, 4, 16), blk, 0, stream>>>(go_w, R, go_b, nullptr, W0, 256, 256, CHW, CHW);
  // 11. glob = LN(glob)
  ln2d_k<<<256, blk, 0, stream>>>(W0, nullptr, gbn_g, gbn_b, W0);
  // 12. sum = local + glob                                   -> W2
  add_k<<<4096, blk, 0, stream>>>(W1, W0, W2, T);
  // 13. fused = x + fuse_pw(sum)                             -> W1
  gemm1x1_k<<<dim3(32, 4, 16), blk, 0, stream>>>(fp_w, W2, fp_b, x, W1, 256, 256, CHW, CHW);
  // 14. xf = LN(fused, fusion_res_norm)                      -> W0
  ln2d_k<<<256, blk, 0, stream>>>(W1, nullptr, frn_g, frn_b, W0);
  // 15. y1 = LN(xf, ffn_norm)                                -> W2
  ln2d_k<<<256, blk, 0, stream>>>(W0, nullptr, ffn_g, ffn_b, W2);
  // 16-18. FFN in two 8-batch chunks (hidden tensors staged in R)
  const long HBS = (long)HIDC * HWSP;   // per-batch hidden elems
  for (int ch = 0; ch < 2; ++ch) {
    const float* y1 = W2 + (long)ch * 8 * CHW;
    float* h1 = R;                 // 8 * HBS = 2T
    float* h2 = R + 2 * T;         // 8 * HBS = 2T
    gemm1x1_k<<<dim3(32, 16, 8), blk, 0, stream>>>(fe_w, y1, fe_b, nullptr, h1, 1024, 256, CHW, HBS);
    dw_k<<<dim3(4, 4, 8 * HIDC), dwblk, 0, stream>>>(h1, HBS, fd3_w, fd3_b, nullptr, nullptr, h2, HBS, HIDC, 0, 1);
    gemm1x1_k<<<dim3(32, 4, 8), blk, 0, stream>>>(fs_w, h2, fs_b, nullptr, W1 + (long)ch * 8 * CHW, 256, 1024, HBS, CHW);
  }
  // 19. out = LN(xf + y, output_norm)                        -> d_out
  ln2d_k<<<256, blk, 0, stream>>>(W0, W1, on_g, on_b, out);
  (void)in_sizes; (void)n_in; (void)out_size; (void)ws_size;
}
